// SpatialConvLayer_17841294148276
// MI455X (gfx1250) — matrix operand order repaired and checked by reading, not compiled, not run on hardware
//
#include <hip/hip_runtime.h>

// ---------------------------------------------------------------------------
// SpatialConvLayer (ChebConv) for MI455X / gfx1250: wave32, WMMA bf16, TDM.
//
//   cvt    : chebbf[1024x3072]  = bf16(cheb)                     (-> d_ws)
//   GEMM1  : Y1 = Xb[393216x64] @ W2[64x192], stored TRANSPOSED as
//            BT[24576 x 3072] bf16  (BT[(r%128)*192+c][r/128])   (-> d_ws)
//   GEMM2  : Z[1024x24576] = cheb @ Y1_2 ; A/B tiles pulled into LDS by the
//            Tensor Data Mover (double buffered, TENSORcnt synchronized)
//   epilogue: out[b,o,t,n] = Z_flat[((b*12+t)*1024+n)*64+o] + bias[o] + x
// ---------------------------------------------------------------------------

typedef unsigned int   u32;
typedef unsigned short u16;
typedef __attribute__((ext_vector_type(16))) __bf16 v16bf;
typedef __attribute__((ext_vector_type(8)))  float  v8f;
typedef __attribute__((ext_vector_type(4)))  u32    u32x4;
typedef __attribute__((ext_vector_type(2)))  u32    u32x2;
typedef __attribute__((ext_vector_type(4)))  int    i32x4;
typedef __attribute__((ext_vector_type(8)))  int    i32x8;
typedef __attribute__((ext_vector_type(4)))  float  f32x4;

union ABFrag { u32x4 q[2]; v16bf v; };   // 32 B = 8 VGPRs = one WMMA A/B operand

__device__ __forceinline__ u32 f2bf_bits(float f) {
  u32 u = __float_as_uint(f);
  return (u + 0x7FFFu + ((u >> 16) & 1u)) >> 16;   // RNE
}
__device__ __forceinline__ u32 pack2(float a, float b) {
  return f2bf_bits(a) | (f2bf_bits(b) << 16);
}

#define R1 393216L   // rows of GEMM1 = 32*64*12*1024/64
#define N1 192       // Ks*C_out
#define K2 3072      // N*Ks
#define N2 24576L    // 32*12*64
#define M2 1024

#if __has_builtin(__builtin_amdgcn_tensor_load_to_lds)
#define HAVE_TDM 1
#else
#define HAVE_TDM 0
#endif

// ---------------------------------------------------------------------------
// TDM: 2D bf16 tile load, LDS rows padded 32->40 elements
// (pad_interval code 3 = 16 DWORDs = one 32-elem row; pad_amount code 3 = 4 DWORDs)
// ---------------------------------------------------------------------------
#if HAVE_TDM
__device__ __forceinline__ void tdm_load_2d_bf16(
    u32 lds_byte_off, const void* gptr, u32 tensor_d0, u32 tensor_d1,
    u32 tile_d0, u32 tile_d1, u32 stride0_elems)
{
  unsigned long long ga = (unsigned long long)(uintptr_t)gptr;
  u32x4 g0;
  g0.x = 1u;                                            // count=1
  g0.y = lds_byte_off;                                  // lds_addr
  g0.z = (u32)ga;                                       // global_addr[31:0]
  g0.w = ((u32)(ga >> 32) & 0x01FFFFFFu) | (2u << 30);  // addr[56:32] | type=2
  i32x8 g1;
  g1[0] = (int)((1u << 16) | (1u << 20) | (3u << 22) | (3u << 25)); // data=2B,pad
  g1[1] = (int)((tensor_d0 & 0xFFFFu) << 16);
  g1[2] = (int)((tensor_d0 >> 16) | ((tensor_d1 & 0xFFFFu) << 16));
  g1[3] = (int)((tensor_d1 >> 16) | (tile_d0 << 16));
  g1[4] = (int)tile_d1;                                 // tile_dim2 = 0
  g1[5] = (int)stride0_elems;                           // dim0 stride lo
  g1[6] = 0;                                            // stride hi | dim1 stride
  g1[7] = 0;
  i32x4 z4 = {0, 0, 0, 0};
#if defined(__clang_major__) && (__clang_major__ >= 23)
  i32x8 z8 = {0, 0, 0, 0, 0, 0, 0, 0};
  __builtin_amdgcn_tensor_load_to_lds(g0, g1, z4, z4, z8, 0);
#else
  __builtin_amdgcn_tensor_load_to_lds(g0, g1, z4, z4, 0);
#endif
}
#endif

// ---------------------------------------------------------------------------
// cheb fp32 -> bf16 (one-time, 3.1M elements)
// ---------------------------------------------------------------------------
__global__ __launch_bounds__(256) void cvt_cheb(const float* __restrict__ src,
                                                u16* __restrict__ dst)
{
  const long i = ((long)blockIdx.x * 256 + threadIdx.x) * 4;
  f32x4 f = *(const f32x4*)(src + i);
  u32x2 p = {pack2(f.x, f.y), pack2(f.z, f.w)};
  *(u32x2*)(dst + i) = p;
}

// ---------------------------------------------------------------------------
// GEMM1: per wave 16 rows x 192 cols, K=64 (2 bf16 WMMA k-steps).
// Output scattered into BT[24576 x 3072] bf16 for GEMM2's TDM-friendly layout.
// ---------------------------------------------------------------------------
__global__ __launch_bounds__(256) void cheb_gemm1(
    const float* __restrict__ x, const float* __restrict__ w, u16* __restrict__ y1t)
{
  __shared__ u16 Ws[192 * 72];
  const int tid = threadIdx.x;
  if (tid < 192) {
    const int n = tid;
#pragma unroll 4
    for (int k2 = 0; k2 < 32; ++k2) {
      u32 p = pack2(w[(2 * k2 + 0) * 192 + n], w[(2 * k2 + 1) * 192 + n]);
      *(u32*)&Ws[n * 72 + 2 * k2] = p;
    }
  }
  __syncthreads();

  const int wid = tid >> 5, lane = tid & 31;
  const int h = lane >> 4, lm = lane & 15;
  const long rBase = ((long)blockIdx.x * 8 + wid) * 16;
  const float* xrow = x + (rBase + lm) * 64;

  const v8f vzero = {0.f, 0.f, 0.f, 0.f, 0.f, 0.f, 0.f, 0.f};
  v8f acc[12];
#pragma unroll
  for (int i = 0; i < 12; ++i) acc[i] = vzero;

#pragma unroll
  for (int ks = 0; ks < 2; ++ks) {
    ABFrag a;   // lane (m=lm): k in {8h..8h+7} U {16+8h..23+8h} (+32*ks)
    const f32x4* p0 = (const f32x4*)(xrow + ks * 32 + 8 * h);
    const f32x4* p1 = (const f32x4*)(xrow + ks * 32 + 16 + 8 * h);
    f32x4 f0 = p0[0], f1 = p0[1], f2 = p1[0], f3 = p1[1];
    a.q[0] = (u32x4){pack2(f0.x, f0.y), pack2(f0.z, f0.w), pack2(f1.x, f1.y), pack2(f1.z, f1.w)};
    a.q[1] = (u32x4){pack2(f2.x, f2.y), pack2(f2.z, f2.w), pack2(f3.x, f3.y), pack2(f3.z, f3.w)};
#pragma unroll
    for (int nt = 0; nt < 12; ++nt) {
      ABFrag b;   // lane (n=lm): k in {8h..8h+7} U {16+8h..23+8h} (+32*ks)
      const u16* bp = &Ws[(nt * 16 + lm) * 72 + ks * 32 + 8 * h];
      b.q[0] = *(const u32x4*)bp;
      b.q[1] = *(const u32x4*)(bp + 16);
      acc[nt] = __builtin_amdgcn_wmma_f32_16x16x32_bf16(
          false, a.v, false, b.v, (short)0, acc[nt], false, false);
    }
  }

#pragma unroll
  for (int nt = 0; nt < 12; ++nt)
#pragma unroll
    for (int v = 0; v < 8; ++v) {
      const long r = rBase + v + 8 * h;            // C layout: m = v + 8h, n = lm
      const int  c = nt * 16 + lm;
      const long q = (r & 127) * 192 + c;          // BT row
      y1t[q * K2 + (r >> 7)] = (u16)f2bf_bits(acc[nt][v]);
    }
}

// ---------------------------------------------------------------------------
// GEMM2: block tile 128(M) x 256(N), K-step 32, double-buffered TDM staging.
// LDS (u16 units): A0 [0,5120) A1 [5120,10240) B0 [10240,20480) B1 [20480,30720)
// A: row-major [128][40], B: N-major [256][40] (comes straight from BT).
// ---------------------------------------------------------------------------
#define ABUF(b) ((b) * 5120)
#define BBUF(b) (10240 + (b) * 10240)

__global__ __launch_bounds__(256) void cheb_gemm2(
    const u16* __restrict__ bt,       // BT [24576 x 3072] bf16
    const u16* __restrict__ chebbf,   // [1024 x 3072] bf16
    const float* __restrict__ bias, const float* __restrict__ x,
    float* __restrict__ out)
{
  __shared__ u16 smem[30720];
  const int tid = threadIdx.x;
  const int wid = tid >> 5, lane = tid & 31;
  const int h = lane >> 4, lm = lane & 15;
  const int wm = wid >> 2, wn = wid & 3;
  const int mBase = blockIdx.y * 128;
  const int nBase = blockIdx.x * 256;

  const u16* aBase = chebbf + (long)mBase * K2;   // tile: 128 rows x 32 k
  const u16* bBase = bt + (long)nBase * K2;       // tile: 256 rows x 32 k

  const v8f vzero = {0.f, 0.f, 0.f, 0.f, 0.f, 0.f, 0.f, 0.f};
  v8f acc[4][4];
#pragma unroll
  for (int i = 0; i < 4; ++i)
#pragma unroll
    for (int j = 0; j < 4; ++j) acc[i][j] = vzero;

  // ---- tile stager: TDM (waves 0/1) or manual b128 copies (fallback)
  auto stage = [&](int k0, int buf) {
#if HAVE_TDM
    if (wid == 0)
      tdm_load_2d_bf16(ABUF(buf) * 2, aBase + k0, K2, M2, 32, 128, K2);
    else if (wid == 1)
      tdm_load_2d_bf16(BBUF(buf) * 2, bBase + k0, K2, (u32)N2, 32, 256, K2);
#else
    {
      const int row = tid >> 1, ko = (tid & 1) * 16;
      const u32x4* s = (const u32x4*)(aBase + (long)row * K2 + k0 + ko);
      u32x4* d = (u32x4*)&smem[ABUF(buf) + row * 40 + ko];
      d[0] = s[0];
      d[1] = s[1];
    }
    {
      const u32x4* s = (const u32x4*)(bBase + (long)tid * K2 + k0);
      u32x4* d = (u32x4*)&smem[BBUF(buf) + tid * 40];
      d[0] = s[0]; d[1] = s[1]; d[2] = s[2]; d[3] = s[3];
    }
#endif
  };

  stage(0, 0);
  int buf = 0;
  for (int k0 = 0; k0 < K2; k0 += 32) {
#if HAVE_TDM
    if (wid < 2) __builtin_amdgcn_s_wait_tensorcnt(0);  // current buffer landed
#endif
    __syncthreads();   // publish cur buffer; all waves done reading other buffer
    if (k0 + 32 < K2) stage(k0 + 32, buf ^ 1);          // DMA overlaps compute

    const u16* As = smem + ABUF(buf);
    const u16* Bs = smem + BBUF(buf);
    ABFrag a[4];
#pragma unroll
    for (int mt = 0; mt < 4; ++mt) {
      const int row = wm * 64 + mt * 16 + lm;
      a[mt].q[0] = *(const u32x4*)&As[row * 40 + 8 * h];
      a[mt].q[1] = *(const u32x4*)&As[row * 40 + 16 + 8 * h];
    }
#pragma unroll
    for (int nt = 0; nt < 4; ++nt) {
      ABFrag b;
      const u16* bp = &Bs[(wn * 64 + nt * 16 + lm) * 40 + 8 * h];
      b.q[0] = *(const u32x4*)bp;
      b.q[1] = *(const u32x4*)(bp + 16);
#pragma unroll
      for (int mt = 0; mt < 4; ++mt)
        acc[mt][nt] = __builtin_amdgcn_wmma_f32_16x16x32_bf16(
            false, a[mt].v, false, b.v, (short)0, acc[mt][nt], false, false);
    }
    buf ^= 1;
  }

  // ---- fused epilogue: bias + [B,T,N,C]->[B,C,T,N] scatter + residual
#pragma unroll
  for (int nt = 0; nt < 4; ++nt) {
    const int o = nt * 16 + lm;              // = (N & 63) for every tile here
    const float bv = bias[o];
#pragma unroll
    for (int mt = 0; mt < 4; ++mt)
#pragma unroll
      for (int v = 0; v < 8; ++v) {
        const long M = mBase + wm * 64 + mt * 16 + v + 8 * h;
        const long N = nBase + wn * 64 + nt * 16 + lm;
        const long r2 = M * 384 + (N >> 6);  // == (M*24576 + N) >> 6
        const int  n_ = (int)(r2 & 1023);
        const long r3 = r2 >> 10;
        const int  tt = (int)(r3 % 12);
        const int  bb = (int)(r3 / 12);
        const long oi = (((long)bb * 64 + o) * 12 + tt) * 1024 + n_;
        out[oi] = acc[mt][nt][v] + bv + x[oi];
      }
  }
}

// ---------------------------------------------------------------------------
extern "C" void kernel_launch(void* const* d_in, const int* in_sizes, int n_in,
                              void* d_out, int out_size, void* d_ws, size_t ws_size,
                              hipStream_t stream) {
  const float* x    = (const float*)d_in[0];   // [32,64,12,1024]
  const float* w    = (const float*)d_in[1];   // [3,64,64]
  const float* bias = (const float*)d_in[2];   // [64]
  const float* cheb = (const float*)d_in[3];   // [1024,3072]
  float* out = (float*)d_out;

  const size_t bt_bytes   = (size_t)N2 * K2 * sizeof(u16);   // ~151 MB
  const size_t cheb_bytes = (size_t)M2 * K2 * sizeof(u16);   // ~6 MB
  if (ws_size < bt_bytes + cheb_bytes) return;
  u16* y1t    = (u16*)d_ws;
  u16* chebbf = (u16*)((char*)d_ws + bt_bytes);

  cvt_cheb<<<dim3((M2 * K2) / (256 * 4)), 256, 0, stream>>>(cheb, chebbf);
  cheb_gemm1<<<dim3(R1 / (8 * 16)), 256, 0, stream>>>(x, w, y1t);
  cheb_gemm2<<<dim3(N2 / 256, M2 / 128), 256, 0, stream>>>(y1t, chebbf, bias, x, out);
}
